// PointNetPlusPlus_38895223832812
// MI455X (gfx1250) — matrix operand order repaired and checked by reading, not compiled, not run on hardware
//
#include <hip/hip_runtime.h>
#include <cmath>

typedef _Float16 h16;
typedef _Float16 v16h __attribute__((ext_vector_type(16)));
typedef float    v8f  __attribute__((ext_vector_type(8)));

#define DEV __device__ __forceinline__

// ---------------- WMMA fragment loaders (CDNA5 ISA 7.12.2 layouts) ----------
// A (16x32 f16, MxK): lane<16 -> row=lane, K in {0..7, 16..23}
//                     lane>=16 -> row=lane-16, K in {8..15, 24..31}
DEV v16h load_frag_A(const h16* base, int lane, int ld) {
  int row = lane & 15;
  int sel = lane >> 4;
  const h16* p = base + row * ld + sel * 8;
  union { uint4 u[2]; v16h v; } r;
  r.u[0] = *(const uint4*)(p);
  r.u[1] = *(const uint4*)(p + 16);
  return r.v;
}
// B (32x16 f16, KxN) sourced from B^T row-major [N][K]:
// lane<16 -> col=lane, K in {0..7, 16..23}; lane>=16 -> col=lane-16, K in {8..15, 24..31}
DEV v16h load_frag_B(const h16* base, int lane, int ld) {
  int row = lane & 15;
  int sel = lane >> 4;
  const h16* p = base + row * ld + sel * 8;
  union { uint4 u[2]; v16h v; } r;
  r.u[0] = *(const uint4*)(p);
  r.u[1] = *(const uint4*)(p + 16);
  return r.v;
}

// ---------------- weight fp32 -> padded f16 ---------------------------------
__global__ void k_cvt(const float* __restrict__ W, int O, int C, int Cp,
                      h16* __restrict__ out) {
  int t = blockIdx.x * 256 + threadIdx.x;
  int tot = O * Cp;
  if (t >= tot) return;
  int o = t / Cp, c = t % Cp;
  out[t] = (c < C) ? (h16)W[(size_t)o * C + c] : (h16)0.f;
}

// ---------------- GEMM: Y[o][m] = act(s[o]*(W.X)[o][m] + b[o]) --------------
// W f16 [O][Cpad], X f16 [Cpad][M], Y f16 [ceil32(O)][M] (pad rows zeroed)
__global__ __launch_bounds__(256)
void k_gemm(const h16* __restrict__ Wh, int O, int Cpad,
            const h16* __restrict__ X, int M,
            const float* __restrict__ sc, const float* __restrict__ bi,
            int act, h16* __restrict__ Y,
            float* __restrict__ f32o, int Ot, int Np) {
  __shared__ h16 Wt[32][32];
  __shared__ h16 Xt[64][32];
  int tid  = threadIdx.x;
  int lane = tid & 31;
  int w    = tid >> 5;
  int m0   = blockIdx.x * 64;
  int ob   = blockIdx.y * 32;
  int o_sub = (w & 1) * 16;
  int m_sub = (w >> 1) * 16;
  v8f acc = {0.f,0.f,0.f,0.f,0.f,0.f,0.f,0.f};
  for (int c0 = 0; c0 < Cpad; c0 += 32) {
    { // stage W tile [32 o][32 c]
      int r = tid >> 3, cg = (tid & 7) * 4;
      int o = ob + r;
      uint2 v = make_uint2(0u, 0u);
      if (o < O) v = *(const uint2*)(Wh + (size_t)o * Cpad + c0 + cg);
      *(uint2*)(&Wt[r][cg]) = v;
    }
    { // stage X tile transposed -> Xt[m][c]
      int c = tid >> 3, mi0 = (tid & 7) * 8;
      union { uint4 u; h16 h[8]; } v;
      v.u = *(const uint4*)(X + (size_t)(c0 + c) * M + m0 + mi0);
#pragma unroll
      for (int i = 0; i < 8; i++) Xt[mi0 + i][c] = v.h[i];
    }
    if (c0 + 32 < Cpad) {
      const h16* nx = X + (size_t)(c0 + 32 + (tid >> 3)) * M + m0 + (tid & 7) * 8;
      __builtin_prefetch((const void*)nx, 0, 1);
    }
    __syncthreads();
    v16h a = load_frag_A(&Wt[o_sub][0], lane, 32);
    v16h b = load_frag_B(&Xt[m_sub][0], lane, 32);
    acc = __builtin_amdgcn_wmma_f32_16x16x32_f16(false, a, false, b,
                                                 (short)0, acc, false, false);
    __syncthreads();
  }
  int mcol = m0 + m_sub + (lane & 15);
  int rowb = ob + o_sub + ((lane >> 4) ? 8 : 0);
#pragma unroll
  for (int e = 0; e < 8; e++) {
    int o = rowb + e;
    float v = 0.f;
    if (o < O) {
      v = acc[e];
      if (sc) v *= sc[o];
      if (bi) v += bi[o];
      if (act == 1) v = v > 0.f ? v : 0.f;
      else if (act == 2) v = 0.5f * v * (1.f + erff(v * 0.70710678f));
    }
    Y[(size_t)o * M + mcol] = (h16)v;
    if (f32o && o < O) {
      int bb = mcol / Np, n = mcol % Np;
      f32o[((size_t)bb * Ot + o) * Np + n] = v;
    }
  }
}

// ---------------- farthest point sampling -----------------------------------
__global__ __launch_bounds__(256)
void k_fps(const float* __restrict__ xyz, int* __restrict__ out, int N, int S) {
  int b = blockIdx.x, tid = threadIdx.x;
  __shared__ float dist[2048];
  __shared__ float rv[256];
  __shared__ int   ri[256];
  __shared__ float fpt[3];
  __shared__ int   far;
  for (int i = tid; i < N; i += 256) dist[i] = 1e10f;
  if (tid == 0) far = 0;
  __syncthreads();
  for (int it = 0; it < S; it++) {
    if (tid == 0) {
      out[b * S + it] = far;
      const float* p = xyz + ((size_t)b * N + far) * 3;
      fpt[0] = p[0]; fpt[1] = p[1]; fpt[2] = p[2];
    }
    __syncthreads();
    float bv = -1.f; int bi2 = 0;
    for (int i = tid; i < N; i += 256) {
      const float* p = xyz + ((size_t)b * N + i) * 3;
      float dx = p[0]-fpt[0], dy = p[1]-fpt[1], dz = p[2]-fpt[2];
      float d = dx*dx + dy*dy + dz*dz;
      float nd = fminf(dist[i], d);
      dist[i] = nd;
      if (nd > bv) { bv = nd; bi2 = i; }
    }
    rv[tid] = bv; ri[tid] = bi2;
    __syncthreads();
    for (int st = 128; st > 0; st >>= 1) {
      if (tid < st && rv[tid + st] > rv[tid]) { rv[tid] = rv[tid+st]; ri[tid] = ri[tid+st]; }
      __syncthreads();
    }
    if (tid == 0) far = ri[0];
    __syncthreads();
  }
}

__global__ void k_gatherxyz(const float* __restrict__ xyz, int N,
                            const int* __restrict__ idx, int S,
                            float* __restrict__ out, int total) {
  int g = blockIdx.x * 256 + threadIdx.x;
  if (g >= total) return;
  int b = g / S;
  int j = idx[g];
  const float* p = xyz + ((size_t)b * N + j) * 3;
  out[(size_t)g*3] = p[0]; out[(size_t)g*3+1] = p[1]; out[(size_t)g*3+2] = p[2];
}

__global__ void k_ball(const float* __restrict__ xyz, const float* __restrict__ nx,
                       int* __restrict__ gi, float r2, int K, int N, int S, int total) {
  int g = blockIdx.x * 256 + threadIdx.x;
  if (g >= total) return;
  int b = g / S;
  const float* c = nx + (size_t)g * 3;
  int* o = gi + (size_t)g * K;
  int cnt = 0, first = -1;
  for (int j = 0; j < N && cnt < K; j++) {
    const float* p = xyz + ((size_t)b * N + j) * 3;
    float dx = p[0]-c[0], dy = p[1]-c[1], dz = p[2]-c[2];
    if (dx*dx + dy*dy + dz*dz <= r2) { if (first < 0) first = j; o[cnt++] = j; }
  }
  if (first < 0) first = N - 1;
  for (; cnt < K; cnt++) o[cnt] = first;
}

// grouped input builder: rows [0..Cpts) = gathered pts (or abs xyz), rows
// [Cpts..Cpts+3) = relative xyz, rest zero.  m = (b*S+s)*K + k
__global__ void k_group(const float* __restrict__ xyz, const float* __restrict__ nx,
                        const h16* __restrict__ pts, int Cpts, int Mpts,
                        const int* __restrict__ gi, int S, int K, int Npts,
                        h16* __restrict__ Xo, int Cpad, int Mtot) {
  int m = blockIdx.x * 256 + threadIdx.x;
  if (m >= Mtot) return;
  int k  = m % K;
  int bs = m / K;
  int b  = bs / S;
  int j  = gi[(size_t)bs * K + k];
  const float* pj = xyz + ((size_t)b * Npts + j) * 3;
  const float* c  = nx + (size_t)bs * 3;
  int D;
  if (pts) {
    D = Cpts;
    for (int ch = 0; ch < Cpts; ch++)
      Xo[(size_t)ch * Mtot + m] = pts[(size_t)ch * Mpts + (size_t)b * Npts + j];
  } else {
    D = 3;
    Xo[m] = (h16)pj[0];
    Xo[(size_t)Mtot + m] = (h16)pj[1];
    Xo[(size_t)2 * Mtot + m] = (h16)pj[2];
  }
  Xo[(size_t)(D+0) * Mtot + m] = (h16)(pj[0] - c[0]);
  Xo[(size_t)(D+1) * Mtot + m] = (h16)(pj[1] - c[1]);
  Xo[(size_t)(D+2) * Mtot + m] = (h16)(pj[2] - c[2]);
  for (int ch = D + 3; ch < Cpad; ch++) Xo[(size_t)ch * Mtot + m] = (h16)0.f;
}

__global__ void k_maxpool(const h16* __restrict__ X, int Min, int K,
                          h16* __restrict__ Y, int Mout, int rows) {
  int t = blockIdx.x * 256 + threadIdx.x;
  if (t >= rows * Mout) return;
  int r = t / Mout, bs = t % Mout;
  const h16* p = X + (size_t)r * Min + (size_t)bs * K;
  float mv = -1e30f;
  for (int k = 0; k < K; k++) mv = fmaxf(mv, (float)p[k]);
  Y[(size_t)r * Mout + bs] = (h16)mv;
}

__global__ void k_xyzrows(const float* __restrict__ xyz, h16* __restrict__ dst, int M) {
  int m = blockIdx.x * 256 + threadIdx.x;
  if (m >= M) return;
  dst[m]                     = (h16)xyz[(size_t)m * 3 + 0];
  dst[(size_t)M + m]         = (h16)xyz[(size_t)m * 3 + 1];
  dst[(size_t)2 * M + m]     = (h16)xyz[(size_t)m * 3 + 2];
}

__global__ void k_bcast(const h16* __restrict__ src, int Bcols, int Sl,
                        h16* __restrict__ dst, int rows, int M) {
  int t = blockIdx.x * 256 + threadIdx.x;
  if (t >= rows * M) return;
  int r = t / M, m = t % M, b = m / Sl;
  dst[t] = src[(size_t)r * Bcols + b];
}

__global__ void k_knn3(const float* __restrict__ x1, const float* __restrict__ x2,
                       int n1, int n2, int* __restrict__ i3, float* __restrict__ w3,
                       int total) {
  int g = blockIdx.x * 256 + threadIdx.x;
  if (g >= total) return;
  int b = g / n1;
  const float* p = x1 + (size_t)g * 3;
  float d0 = 1e30f, d1 = 1e30f, d2 = 1e30f;
  int j0 = 0, j1 = 0, j2 = 0;
  for (int j = 0; j < n2; j++) {
    const float* q = x2 + ((size_t)b * n2 + j) * 3;
    float dx = p[0]-q[0], dy = p[1]-q[1], dz = p[2]-q[2];
    float d = dx*dx + dy*dy + dz*dz;
    if (d < d0)      { d2=d1; j2=j1; d1=d0; j1=j0; d0=d; j0=j; }
    else if (d < d1) { d2=d1; j2=j1; d1=d;  j1=j; }
    else if (d < d2) { d2=d;  j2=j; }
  }
  float w0 = 1.f/(d0+1e-8f), w1 = 1.f/(d1+1e-8f), w2 = 1.f/(d2+1e-8f);
  float s = w0 + w1 + w2;
  i3[(size_t)g*3]=j0; i3[(size_t)g*3+1]=j1; i3[(size_t)g*3+2]=j2;
  w3[(size_t)g*3]=w0/s; w3[(size_t)g*3+1]=w1/s; w3[(size_t)g*3+2]=w2/s;
}

__global__ void k_interp(const h16* __restrict__ src, int C2, int Min, int n2,
                         const int* __restrict__ i3, const float* __restrict__ w3,
                         h16* __restrict__ dst, int Mout, int n1) {
  int m = blockIdx.x * 256 + threadIdx.x;
  if (m >= Mout) return;
  int b = m / n1;
  int j0 = i3[(size_t)m*3], j1 = i3[(size_t)m*3+1], j2 = i3[(size_t)m*3+2];
  float w0 = w3[(size_t)m*3], w1 = w3[(size_t)m*3+1], w2 = w3[(size_t)m*3+2];
  for (int c = 0; c < C2; c++) {
    const h16* row = src + (size_t)c * Min + (size_t)b * n2;
    dst[(size_t)c * Mout + m] =
        (h16)(w0*(float)row[j0] + w1*(float)row[j1] + w2*(float)row[j2]);
  }
}

__global__ void k_text(const float* __restrict__ tf, const float* __restrict__ Wt,
                       const float* __restrict__ bt, float* __restrict__ tv) {
  int t = blockIdx.x * 256 + threadIdx.x;
  if (t >= 8 * 128) return;
  int b = t / 128, o = t % 128;
  float a = bt[o];
  const float* wr = Wt + (size_t)o * 512;
  const float* xr = tf + (size_t)b * 512;
  for (int c = 0; c < 512; c++) a += wr[c] * xr[c];
  tv[t] = 0.5f * a * (1.f + erff(a * 0.70710678f));
}

__global__ void k_fuse(const h16* __restrict__ pf, const float* __restrict__ tv,
                       h16* __restrict__ out, int M, int n) {
  int t = blockIdx.x * 256 + threadIdx.x;
  if (t >= 128 * M) return;
  int c = t / M, m = t % M, b = m / n;
  out[t] = (h16)((float)pf[t] + tv[b * 128 + c]);
}

// ---------------- flash attention: one wave per (b, head, 16 queries) -------
__global__ __launch_bounds__(32)
void k_attn(const h16* __restrict__ q, const h16* __restrict__ k,
            const h16* __restrict__ v, h16* __restrict__ out, int M, int n) {
  int lane = threadIdx.x;
  int qt = blockIdx.x * 16;
  int h  = blockIdx.y;
  int b  = blockIdx.z;
  size_t base = (size_t)b * n;
  __shared__ h16 Qld[16][32];
  __shared__ h16 Kld[32][32];   // [key][d]
  __shared__ h16 Vld[32][32];   // [d][key]
  __shared__ float Sld[16][32];
  __shared__ h16 Pld[16][32];
  __shared__ float rowm[16], rowl[16], alf[16];
  for (int t = lane; t < 16 * 32; t += 32) {
    int r = t & 15, d = t >> 4;
    Qld[r][d] = q[(size_t)(h * 32 + d) * M + base + qt + r];
  }
  if (lane < 16) { rowm[lane] = -1e30f; rowl[lane] = 0.f; }
  __syncthreads();
  v16h af = load_frag_A(&Qld[0][0], lane, 32);
  v8f o0 = {0.f,0.f,0.f,0.f,0.f,0.f,0.f,0.f};
  v8f o1 = {0.f,0.f,0.f,0.f,0.f,0.f,0.f,0.f};
  for (int kb = 0; kb < n; kb += 32) {
    for (int t = lane; t < 32 * 32; t += 32) {
      int kl = t & 31, d = t >> 5;
      size_t gi = (size_t)(h * 32 + d) * M + base + kb + kl;
      Kld[kl][d] = k[gi];
      Vld[d][kl] = v[gi];
    }
    __syncthreads();
    v8f s0 = {0.f,0.f,0.f,0.f,0.f,0.f,0.f,0.f};
    v8f s1 = {0.f,0.f,0.f,0.f,0.f,0.f,0.f,0.f};
    v16h b0 = load_frag_B(&Kld[0][0], lane, 32);
    v16h b1 = load_frag_B(&Kld[16][0], lane, 32);
    s0 = __builtin_amdgcn_wmma_f32_16x16x32_f16(false, af, false, b0, (short)0, s0, false, false);
    s1 = __builtin_amdgcn_wmma_f32_16x16x32_f16(false, af, false, b1, (short)0, s1, false, false);
    {
      int nc = lane & 15, rb = (lane >> 4) * 8;
#pragma unroll
      for (int e = 0; e < 8; e++) { Sld[rb+e][nc] = s0[e]; Sld[rb+e][16+nc] = s1[e]; }
    }
    __syncthreads();
    if (lane < 16) {
      int r = lane;
      float mold = rowm[r], mnew = mold;
      for (int j = 0; j < 32; j++) mnew = fmaxf(mnew, Sld[r][j]);
      float al = __expf(mold - mnew);
      float l = rowl[r] * al;
      for (int j = 0; j < 32; j++) {
        float p = __expf(Sld[r][j] - mnew);
        Pld[r][j] = (h16)p;
        l += p;
      }
      rowm[r] = mnew; rowl[r] = l; alf[r] = al;
    }
    __syncthreads();
    {
      int rb = (lane >> 4) * 8;
#pragma unroll
      for (int e = 0; e < 8; e++) { float a = alf[rb + e]; o0[e] *= a; o1[e] *= a; }
    }
    v16h pfr = load_frag_A(&Pld[0][0], lane, 32);
    v16h vb0 = load_frag_B(&Vld[0][0], lane, 32);
    v16h vb1 = load_frag_B(&Vld[16][0], lane, 32);
    o0 = __builtin_amdgcn_wmma_f32_16x16x32_f16(false, pfr, false, vb0, (short)0, o0, false, false);
    o1 = __builtin_amdgcn_wmma_f32_16x16x32_f16(false, pfr, false, vb1, (short)0, o1, false, false);
    __syncthreads();
  }
  {
    int nc = lane & 15, rb = (lane >> 4) * 8;
#pragma unroll
    for (int e = 0; e < 8; e++) {
      int qq = qt + rb + e;
      float inv = 1.f / rowl[rb + e];
      out[(size_t)(h * 32 + nc) * M + base + qq]      = (h16)(o0[e] * inv);
      out[(size_t)(h * 32 + 16 + nc) * M + base + qq] = (h16)(o1[e] * inv);
    }
  }
}

__global__ void k_gatfinal(const h16* __restrict__ y2, const h16* __restrict__ pf,
                           const float* __restrict__ g, h16* __restrict__ out, int tot) {
  int t = blockIdx.x * 256 + threadIdx.x;
  if (t >= tot) return;
  out[t] = (h16)(g[0] * (float)y2[t] + (float)pf[t]);
}

__global__ void k_writec(const h16* __restrict__ l3, const float* __restrict__ tf,
                         float* __restrict__ out) {
  int t = blockIdx.x * 256 + threadIdx.x;
  if (t >= 8 * 1536) return;
  int b = t / 1536, j = t % 1536;
  out[t] = (j < 1024) ? (float)l3[(size_t)j * 8 + b] : tf[(size_t)b * 512 + (j - 1024)];
}

// ============================================================================
extern "C" void kernel_launch(void* const* d_in, const int* in_sizes, int n_in,
                              void* d_out, int out_size, void* d_ws, size_t ws_size,
                              hipStream_t stream) {
  (void)in_sizes; (void)n_in; (void)out_size; (void)ws_size;
  const int Bn = 8, N0 = 2048, S1 = 512, S2 = 128;
  const int M0 = Bn * N0, M1 = Bn * S1, M2 = Bn * S2;
  const float* xyz = (const float*)d_in[0];
  const float* tf  = (const float*)d_in[1];

  char* base = (char*)d_ws; size_t off = 0;
  auto alloc = [&](size_t bytes) -> void* {
    void* p = base + off; off = (off + bytes + 255) & ~(size_t)255; return p;
  };

  // ---- convert weights (param indices per jax tree_flatten sorted-key order)
  h16* Wh[96] = {};
  auto cvt = [&](int wi, int O, int C) {
    int Cp = (C + 31) & ~31;
    h16* p = (h16*)alloc((size_t)O * Cp * sizeof(h16));
    int tot = O * Cp;
    k_cvt<<<(tot + 255) / 256, 256, 0, stream>>>((const float*)d_in[wi], O, C, Cp, p);
    Wh[wi] = p;
  };
  struct L { int wi, O, C; };
  static const L sa1L[3][3] = {{{36,32,6},{39,32,32},{42,64,32}},
                               {{45,64,6},{48,64,64},{51,128,64}},
                               {{54,64,6},{57,96,64},{60,128,96}}};
  static const L sa2L[2][3] = {{{63,128,323},{66,128,128},{69,256,128}},
                               {{72,128,323},{75,196,128},{78,256,196}}};
  static const L sa3L[3] = {{81,256,515},{84,512,256},{87,1024,512}};
  static const L fp3L[2] = {{17,256,1536},{20,256,256}};
  static const L fp2L[2] = {{11,256,576},{14,128,256}};
  static const L fp1L[2] = {{5,128,134},{8,128,128}};
  for (auto& br : sa1L) for (auto& l : br) cvt(l.wi, l.O, l.C);
  for (auto& br : sa2L) for (auto& l : br) cvt(l.wi, l.O, l.C);
  for (auto& l : sa3L) cvt(l.wi, l.O, l.C);
  for (auto& l : fp3L) cvt(l.wi, l.O, l.C);
  for (auto& l : fp2L) cvt(l.wi, l.O, l.C);
  for (auto& l : fp1L) cvt(l.wi, l.O, l.C);
  cvt(26,128,128); cvt(25,128,128); cvt(28,128,128);  // Wq, Wk, Wv
  cvt(23,128,128); cvt(24,128,128);                   // W1, W2
  cvt(2, 512, 128);                                   // final

  // ---- workspace buffers
  float* nx1  = (float*)alloc((size_t)M1 * 3 * 4);
  float* nx2  = (float*)alloc((size_t)M2 * 3 * 4);
  int*   fidx = (int*)alloc((size_t)Bn * 512 * 4);
  int*   gidx = (int*)alloc((size_t)Bn * 512 * 128 * 4);
  h16* l1p  = (h16*)alloc((size_t)320 * M1 * 2);
  h16* l2p  = (h16*)alloc((size_t)512 * M2 * 2);
  h16* l3p  = (h16*)alloc((size_t)1024 * Bn * 2);
  h16* l2p2 = (h16*)alloc((size_t)256 * M2 * 2);
  h16* l1p2 = (h16*)alloc((size_t)128 * M1 * 2);
  h16* l0p  = (h16*)alloc((size_t)128 * M0 * 2);
  h16* fusedB = (h16*)alloc((size_t)128 * M0 * 2);
  h16* qb   = (h16*)alloc((size_t)128 * M0 * 2);
  h16* kb   = (h16*)alloc((size_t)128 * M0 * 2);
  h16* vb   = (h16*)alloc((size_t)128 * M0 * 2);
  h16* attno= (h16*)alloc((size_t)128 * M0 * 2);
  h16* y1b  = (h16*)alloc((size_t)128 * M0 * 2);
  h16* y2b  = (h16*)alloc((size_t)128 * M0 * 2);
  h16* l0p2 = (h16*)alloc((size_t)128 * M0 * 2);
  h16* yfin = (h16*)alloc((size_t)512 * M0 * 2);
  float* tvec = (float*)alloc((size_t)Bn * 128 * 4);
  int*   i3a = (int*)alloc((size_t)M1 * 3 * 4);
  float* w3a = (float*)alloc((size_t)M1 * 3 * 4);
  int*   i3b = (int*)alloc((size_t)M0 * 3 * 4);
  float* w3b = (float*)alloc((size_t)M0 * 3 * 4);
  h16* bufA = (h16*)alloc((size_t)128 * 524288 * 2);
  h16* bufB = (h16*)alloc((size_t)96 * 524288 * 2);
  h16* bufG = (h16*)alloc((size_t)352 * 131072 * 2);

  auto gemm = [&](const h16* W, int O, int C, const h16* X, int M,
                  const float* s, const float* bi, int act, h16* Y,
                  float* f32o, int Ot, int Np) {
    dim3 g(M / 64, (O + 31) / 32);
    k_gemm<<<g, 256, 0, stream>>>(W, O, (C + 31) & ~31, X, M, s, bi, act, Y, f32o, Ot, Np);
  };
  auto chain = [&](const L* Ls, int n, const h16* X0, int M, h16* pA, h16* pB) -> const h16* {
    const h16* cur = X0;
    for (int i = 0; i < n; i++) {
      h16* o = (i & 1) ? pB : pA;
      gemm(Wh[Ls[i].wi], Ls[i].O, Ls[i].C, cur, M,
           (const float*)d_in[Ls[i].wi + 2], (const float*)d_in[Ls[i].wi + 1],
           1, o, nullptr, 0, 0);
      cur = o;
    }
    return cur;
  };

  // ---- SA1 -----------------------------------------------------------------
  k_fps<<<Bn, 256, 0, stream>>>(xyz, fidx, N0, S1);
  k_gatherxyz<<<(M1 + 255) / 256, 256, 0, stream>>>(xyz, N0, fidx, S1, nx1, M1);
  {
    const float rad[3] = {0.1f, 0.2f, 0.4f};
    const int Kk[3] = {32, 64, 128};
    const int row0[3] = {0, 64, 192};
    for (int br = 0; br < 3; br++) {
      int K = Kk[br], Mg = Bn * S1 * K;
      k_ball<<<(M1 + 255) / 256, 256, 0, stream>>>(xyz, nx1, gidx, rad[br]*rad[br], K, N0, S1, M1);
      k_group<<<(Mg + 255) / 256, 256, 0, stream>>>(xyz, nx1, (const h16*)nullptr, 3, 0,
                                                    gidx, S1, K, N0, bufG, 32, Mg);
      const h16* o = chain(sa1L[br], 3, bufG, Mg, bufA, bufB);
      int rows = sa1L[br][2].O;
      k_maxpool<<<((rows * M1) + 255) / 256, 256, 0, stream>>>(o, Mg, K, l1p + (size_t)row0[br] * M1, M1, rows);
    }
  }
  // ---- SA2 -----------------------------------------------------------------
  k_fps<<<Bn, 256, 0, stream>>>(nx1, fidx, S1, S2);
  k_gatherxyz<<<(M2 + 255) / 256, 256, 0, stream>>>(nx1, S1, fidx, S2, nx2, M2);
  {
    const float rad[2] = {0.4f, 0.8f};
    const int Kk[2] = {64, 128};
    const int row0[2] = {0, 256};
    for (int br = 0; br < 2; br++) {
      int K = Kk[br], Mg = Bn * S2 * K;
      k_ball<<<(M2 + 255) / 256, 256, 0, stream>>>(nx1, nx2, gidx, rad[br]*rad[br], K, S1, S2, M2);
      k_group<<<(Mg + 255) / 256, 256, 0, stream>>>(nx1, nx2, l1p, 320, M1,
                                                    gidx, S2, K, S1, bufG, 352, Mg);
      const h16* o = chain(sa2L[br], 3, bufG, Mg, bufA, bufB);
      k_maxpool<<<((256 * M2) + 255) / 256, 256, 0, stream>>>(o, Mg, K, l2p + (size_t)row0[br] * M2, M2, 256);
    }
  }
  // ---- SA3 (group-all) -----------------------------------------------------
  hipMemsetAsync(bufG, 0, (size_t)544 * M2 * 2, stream);
  k_xyzrows<<<(M2 + 255) / 256, 256, 0, stream>>>(nx2, bufG, M2);
  hipMemcpyAsync(bufG + (size_t)3 * M2, l2p, (size_t)512 * M2 * 2, hipMemcpyDeviceToDevice, stream);
  {
    const h16* o = chain(sa3L, 3, bufG, M2, bufA, bufB);
    k_maxpool<<<((1024 * Bn) + 255) / 256, 256, 0, stream>>>(o, M2, S2, l3p, Bn, 1024);
  }
  // ---- FP3 (broadcast interp) ---------------------------------------------
  hipMemcpyAsync(bufG, l2p, (size_t)512 * M2 * 2, hipMemcpyDeviceToDevice, stream);
  k_bcast<<<((1024 * M2) + 255) / 256, 256, 0, stream>>>(l3p, Bn, S2, bufG + (size_t)512 * M2, 1024, M2);
  chain(fp3L, 2, bufG, M2, bufA, l2p2);
  // ---- FP2 -----------------------------------------------------------------
  k_knn3<<<(M1 + 255) / 256, 256, 0, stream>>>(nx1, nx2, S1, S2, i3a, w3a, M1);
  hipMemcpyAsync(bufG, l1p, (size_t)320 * M1 * 2, hipMemcpyDeviceToDevice, stream);
  k_interp<<<(M1 + 255) / 256, 256, 0, stream>>>(l2p2, 256, M2, S2, i3a, w3a, bufG + (size_t)320 * M1, M1, S1);
  chain(fp2L, 2, bufG, M1, bufA, l1p2);
  // ---- FP1 -----------------------------------------------------------------
  k_knn3<<<(M0 + 255) / 256, 256, 0, stream>>>(xyz, nx1, N0, S1, i3b, w3b, M0);
  hipMemsetAsync(bufG, 0, (size_t)160 * M0 * 2, stream);
  k_xyzrows<<<(M0 + 255) / 256, 256, 0, stream>>>(xyz, bufG, M0);
  k_xyzrows<<<(M0 + 255) / 256, 256, 0, stream>>>(xyz, bufG + (size_t)3 * M0, M0);
  k_interp<<<(M0 + 255) / 256, 256, 0, stream>>>(l1p2, 128, M1, S1, i3b, w3b, bufG + (size_t)6 * M0, M0, N0);
  chain(fp1L, 2, bufG, M0, bufA, l0p);
  // ---- GAT -----------------------------------------------------------------
  k_text<<<((Bn * 128) + 255) / 256, 256, 0, stream>>>(tf, (const float*)d_in[27], (const float*)d_in[33], tvec);
  k_fuse<<<((128 * M0) + 255) / 256, 256, 0, stream>>>(l0p, tvec, fusedB, M0, N0);
  gemm(Wh[26], 128, 128, l0p,    M0, nullptr, (const float*)d_in[32], 0, qb, nullptr, 0, 0);
  gemm(Wh[25], 128, 128, fusedB, M0, nullptr, (const float*)d_in[31], 0, kb, nullptr, 0, 0);
  gemm(Wh[28], 128, 128, l0p,    M0, nullptr, (const float*)d_in[34], 0, vb, nullptr, 0, 0);
  {
    dim3 g(N0 / 16, 4, Bn);
    k_attn<<<g, 32, 0, stream>>>(qb, kb, vb, attno, M0, N0);
  }
  gemm(Wh[23], 128, 128, attno, M0, nullptr, (const float*)d_in[29], 1, y1b, nullptr, 0, 0);
  gemm(Wh[24], 128, 128, y1b,   M0, nullptr, (const float*)d_in[30], 0, y2b, nullptr, 0, 0);
  k_gatfinal<<<((128 * M0) + 255) / 256, 256, 0, stream>>>(y2b, l0p, (const float*)d_in[35], l0p2, 128 * M0);
  // ---- final MLP (gelu) -> f32 d_out in [B,512,N] layout --------------------
  gemm(Wh[2], 512, 128, l0p2, M0, (const float*)d_in[4], (const float*)d_in[3], 2,
       yfin, (float*)d_out, 512, N0);
  // ---- c = concat(l3_points[:, :, 0], text) --------------------------------
  k_writec<<<((Bn * 1536) + 255) / 256, 256, 0, stream>>>(l3p, tf, (float*)d_out + (size_t)Bn * 512 * N0);
}
